// GQAttention_37864431682348
// MI455X (gfx1250) — matrix operand order repaired and checked by reading, not compiled, not run on hardware
//
#include <hip/hip_runtime.h>
#include <hip/hip_bf16.h>

// ---------------------------------------------------------------------------
// GQA attention forward for MI455X (gfx1250): bf16 WMMA + async LDS staging.
// ---------------------------------------------------------------------------

typedef __bf16 bf16_t;
typedef __attribute__((ext_vector_type(16))) __bf16 bf16x16;
typedef __attribute__((ext_vector_type(8)))  __bf16 bf16x8;
typedef __attribute__((ext_vector_type(8)))  float  f32x8;
typedef __attribute__((ext_vector_type(4)))  int    v4i_t;

typedef __attribute__((address_space(1))) v4i_t* as1_v4i;
typedef __attribute__((address_space(3))) v4i_t* as3_v4i;

#define T_SEQ    2048
#define D_MODEL  4096
#define NH_Q     32
#define NH_KV    8
#define HEAD_D   128
#define KV_D     (NH_KV * HEAD_D)   // 1024

// --- async global->LDS path (gfx1250). Guarded so compile never breaks. ----
#if defined(__has_builtin)
#if __has_builtin(__builtin_amdgcn_global_load_async_to_lds_b128)
#define HAVE_ASYNC_LDS 1
#endif
#endif
#ifndef HAVE_ASYNC_LDS
#define HAVE_ASYNC_LDS 0
#endif

// copy 16 bytes global -> LDS (async on gfx1250; sync VGPR fallback otherwise)
__device__ inline void copy16_g2l(const bf16_t* g, bf16_t* l) {
#if HAVE_ASYNC_LDS
  __builtin_amdgcn_global_load_async_to_lds_b128((as1_v4i)(g), (as3_v4i)(l), 0, 0);
#else
  *(bf16x8*)l = *(const bf16x8*)g;
#endif
}

// ---------------------------------------------------------------------------
// 1) elementwise f32 -> bf16 cast
// ---------------------------------------------------------------------------
__global__ void cast_f32_bf16(const float* __restrict__ in, bf16_t* __restrict__ out, int n) {
  int i = blockIdx.x * blockDim.x + threadIdx.x;
  if (i < n) out[i] = (bf16_t)in[i];
}

// ---------------------------------------------------------------------------
// 2) GEMM: C[M,N] = A[M,K] @ B[N,K]^T  with double-buffered async LDS staging.
//    Block = 256 thr (8 waves), tile 128x128, K-step 32.
//    Wave w -> rows [w*16, w*16+16), all 128 cols (8 C fragments, A reused 8x).
//    TRANS=true stores C transposed (C[col*M + row]) -- used to produce V^T.
// ---------------------------------------------------------------------------
#define LSTR 40   // padded LDS row stride (elems): 20 words -> conflict-free

template <typename OutT, bool TRANS>
__global__ void __launch_bounds__(256)
gemm_bf16_nt(const bf16_t* __restrict__ A, const bf16_t* __restrict__ B,
             OutT* __restrict__ C, int M, int N, int K) {
  __shared__ bf16_t lsA[2][128 * LSTR];
  __shared__ bf16_t lsB[2][128 * LSTR];

  const int tid   = threadIdx.x;
  const int lane  = tid & 31;
  const int wave  = tid >> 5;
  const int m0    = blockIdx.x * 128;
  const int n0    = blockIdx.y * 128;
  const int colid = lane & 15;
  const int kA    = (lane >> 4) << 3;   // 0 / 8, unused (A frag also uses kB)
  const int kB    = (lane >> 4) << 4;   // B frag K-offset: 0 / 16

  f32x8 acc[8] = {};

  // stage one 128x32 A-tile + 128x32 B-tile (4 x b128 issues per thread)
  auto stage = [&](int buf, int k0) {
#pragma unroll
    for (int i = 0; i < 2; ++i) {
      const int chunk = tid + i * 256;          // 0..511
      const int row = chunk >> 2, c = chunk & 3;
      copy16_g2l(A + (size_t)(m0 + row) * K + k0 + c * 8,
                 &lsA[buf][row * LSTR + c * 8]);
      copy16_g2l(B + (size_t)(n0 + row) * K + k0 + c * 8,
                 &lsB[buf][row * LSTR + c * 8]);
    }
  };

  stage(0, 0);
  int buf = 0;
  for (int k0 = 0; k0 < K; k0 += 32) {
    const bool more = (k0 + 32) < K;
    if (more) stage(buf ^ 1, k0 + 32);          // prefetch next K-step
#if HAVE_ASYNC_LDS
    if (more) asm volatile("s_wait_asynccnt 0x4" ::: "memory");  // prev stage done
    else      asm volatile("s_wait_asynccnt 0x0" ::: "memory");
#endif
    __syncthreads();

    union { bf16x16 v; bf16x8 h[2]; } a;
    const bf16_t* la = &lsA[buf][(wave * 16 + colid) * LSTR + kB];
    a.h[0] = *(const bf16x8*)la;         // K = kB + 0..7
    a.h[1] = *(const bf16x8*)(la + 8);   // K = kB + 8..15
#pragma unroll
    for (int s = 0; s < 8; ++s) {
      union { bf16x16 v; bf16x8 h[2]; } b;
      const bf16_t* lb = &lsB[buf][(s * 16 + colid) * LSTR + kB];
      b.h[0] = *(const bf16x8*)lb;        // 16 contiguous K
      b.h[1] = *(const bf16x8*)(lb + 8);
      acc[s] = __builtin_amdgcn_wmma_f32_16x16x32_bf16(false, a.v, false, b.v,
                                                       (short)0, acc[s], false, false);
    }
    __syncthreads();
    buf ^= 1;
  }

  const int rb = m0 + wave * 16 + ((lane >> 4) << 3);
#pragma unroll
  for (int s = 0; s < 8; ++s) {
#pragma unroll
    for (int r = 0; r < 8; ++r) {
      const int row = rb + r, col = n0 + s * 16 + colid;
      if (TRANS) C[(size_t)col * M + row] = (OutT)acc[s][r];
      else       C[(size_t)row * N + col] = (OutT)acc[s][r];
    }
  }
}

// ---------------------------------------------------------------------------
// 3) per-head RMSNorm + RoPE (f32 in -> bf16 out). One wave per (t, head).
// ---------------------------------------------------------------------------
__global__ void __launch_bounds__(256)
rmsnorm_rope(const float* __restrict__ in, const float* __restrict__ g,
             const float* __restrict__ cs, const float* __restrict__ sn,
             bf16_t* __restrict__ out, int nheads) {
  const int lane = threadIdx.x & 31;
  const int w    = (blockIdx.x << 3) + (threadIdx.x >> 5);
  const int t    = w / nheads;
  const int h    = w % nheads;
  const size_t row = (size_t)t * nheads * HEAD_D + (size_t)h * HEAD_D;

  float v0 = in[row + lane];
  float v1 = in[row + lane + 32];
  float v2 = in[row + lane + 64];
  float v3 = in[row + lane + 96];

  float ss = v0 * v0 + v1 * v1 + v2 * v2 + v3 * v3;
#pragma unroll
  for (int off = 16; off; off >>= 1) ss += __shfl_xor(ss, off, 32);
  const float rn = rsqrtf(ss * (1.0f / 128.0f) + 1e-6f);

  v0 *= rn * g[lane];
  v1 *= rn * g[lane + 32];
  v2 *= rn * g[lane + 64];
  v3 *= rn * g[lane + 96];

  const float c0 = cs[t * 64 + lane],      s0 = sn[t * 64 + lane];
  const float c1 = cs[t * 64 + lane + 32], s1 = sn[t * 64 + lane + 32];
  out[row + lane]      = (bf16_t)(v0 * c0 - v2 * s0);
  out[row + lane + 32] = (bf16_t)(v1 * c1 - v3 * s1);
  out[row + lane + 64] = (bf16_t)(v2 * c0 + v0 * s0);
  out[row + lane + 96] = (bf16_t)(v3 * c1 + v1 * s1);
}

// ---------------------------------------------------------------------------
// 4) causal flash attention, one wave per (head, 16-query tile).
//    Q: [T, 4096] bf16, K: [T, 1024] bf16, Vt: [1024, T] bf16 (transposed).
//    All 8 K/V fragments of a 32-key step are loaded before the WMMA chain.
// ---------------------------------------------------------------------------
#define P_STRIDE 40

__global__ void __launch_bounds__(256)
flash_attn(const bf16_t* __restrict__ Q, const bf16_t* __restrict__ Kb,
           const bf16_t* __restrict__ Vt, bf16_t* __restrict__ O) {
  __shared__ bf16_t pbuf[8][16 * P_STRIDE];

  const int lane  = threadIdx.x & 31;
  const int wave  = threadIdx.x >> 5;
  const int gw    = blockIdx.x * 8 + wave;   // 0 .. 32*128-1
  const int h     = gw >> 7;                 // head
  const int q0    = (gw & 127) << 4;         // query tile start
  const int kvh   = h >> 2;                  // GQA group
  const int colid = lane & 15;
  const int rbase = (lane >> 4) << 3;
  const int kA    = (lane >> 4) << 3;
  const int kB    = (lane >> 4) << 4;

  const bf16_t* Qh  = Q  + (size_t)h * HEAD_D;                 // row stride 4096
  const bf16_t* Kh  = Kb + (size_t)kvh * HEAD_D;               // row stride 1024
  const bf16_t* Vth = Vt + (size_t)kvh * HEAD_D * T_SEQ;       // row stride T

  bf16x16 qa[4];
  {
    const bf16_t* qp = Qh + (size_t)(q0 + colid) * D_MODEL + kB;
#pragma unroll
    for (int i = 0; i < 4; ++i) {
      union { bf16x16 v; bf16x8 hh[2]; } u;
      u.hh[0] = *(const bf16x8*)(qp + 32 * i);
      u.hh[1] = *(const bf16x8*)(qp + 32 * i + 8);
      qa[i] = u.v;
    }
  }

  f32x8 of[8] = {};
  float m[8], l[8];
#pragma unroll
  for (int r = 0; r < 8; ++r) { m[r] = -1e30f; l[r] = 0.0f; }

  bf16_t* plds = pbuf[wave];
  const float scale = 0.08838834764831845f;  // 1/sqrt(128)

  for (int s0 = 0; s0 < q0 + 16; s0 += 32) {
    // ---- load all 8 K B-fragments first (batched, one wait) --------------
    bf16x16 kf[8];
#pragma unroll
    for (int half = 0; half < 2; ++half) {
      const bf16_t* kp = Kh + (size_t)(s0 + half * 16 + colid) * KV_D + kB;
#pragma unroll
      for (int i = 0; i < 4; ++i) kf[half * 4 + i] = *(const bf16x16*)(kp + 32 * i);
    }
    // ---- scores: two 16x16 fragments (keys s0..s0+31) --------------------
    f32x8 sc[2] = {};
#pragma unroll
    for (int half = 0; half < 2; ++half)
#pragma unroll
      for (int i = 0; i < 4; ++i)
        sc[half] = __builtin_amdgcn_wmma_f32_16x16x32_bf16(
            false, qa[i], false, kf[half * 4 + i], (short)0, sc[half], false, false);

    // ---- scale + causal mask ---------------------------------------------
#pragma unroll
    for (int half = 0; half < 2; ++half) {
      const int kc = s0 + half * 16 + colid;
#pragma unroll
      for (int r = 0; r < 8; ++r) {
        const int qrow = q0 + rbase + r;
        sc[half][r] = (kc <= qrow) ? sc[half][r] * scale : -1e30f;
      }
    }
    // ---- online softmax ---------------------------------------------------
    float p0[8], p1[8], sf[8];
#pragma unroll
    for (int r = 0; r < 8; ++r) {
      float rm = fmaxf(sc[0][r], sc[1][r]);
      rm = fmaxf(rm, __shfl_xor(rm, 1, 32));
      rm = fmaxf(rm, __shfl_xor(rm, 2, 32));
      rm = fmaxf(rm, __shfl_xor(rm, 4, 32));
      rm = fmaxf(rm, __shfl_xor(rm, 8, 32));
      const float mn = fmaxf(m[r], rm);
      sf[r] = __expf(m[r] - mn);
      p0[r] = __expf(sc[0][r] - mn);
      p1[r] = __expf(sc[1][r] - mn);
      float rs = p0[r] + p1[r];
      rs += __shfl_xor(rs, 1, 32);
      rs += __shfl_xor(rs, 2, 32);
      rs += __shfl_xor(rs, 4, 32);
      rs += __shfl_xor(rs, 8, 32);
      l[r] = l[r] * sf[r] + rs;
      m[r] = mn;
    }
#pragma unroll
    for (int n = 0; n < 8; ++n)
#pragma unroll
      for (int r = 0; r < 8; ++r) of[n][r] *= sf[r];

    // ---- P (C-layout) -> LDS -> A-layout fragment ------------------------
#pragma unroll
    for (int r = 0; r < 8; ++r) {
      plds[(rbase + r) * P_STRIDE + colid]      = (bf16_t)p0[r];
      plds[(rbase + r) * P_STRIDE + 16 + colid] = (bf16_t)p1[r];
    }
    asm volatile("s_wait_dscnt 0x0" ::: "memory");  // same-wave LDS store->load
    union { bf16x16 v; bf16x8 hh[2]; } pa;
    pa.hh[0] = *(const bf16x8*)(plds + colid * P_STRIDE + kB);
    pa.hh[1] = *(const bf16x8*)(plds + colid * P_STRIDE + kB + 8);

    // ---- load all 8 V^T B-fragments (contiguous 32B each), then WMMAs ----
    bf16x16 vf[8];
#pragma unroll
    for (int n = 0; n < 8; ++n)
      vf[n] = *(const bf16x16*)(Vth + (size_t)(n * 16 + colid) * T_SEQ + s0 + kB);
#pragma unroll
    for (int n = 0; n < 8; ++n)
      of[n] = __builtin_amdgcn_wmma_f32_16x16x32_bf16(false, pa.v, false, vf[n],
                                                      (short)0, of[n], false, false);
  }

  // ---- normalize and store bf16 attention output [T, 4096] ---------------
  bf16_t* op = O + (size_t)h * HEAD_D;
#pragma unroll
  for (int r = 0; r < 8; ++r) {
    const float inv = 1.0f / l[r];
    const size_t trow = (size_t)(q0 + rbase + r) * D_MODEL;
#pragma unroll
    for (int n = 0; n < 8; ++n)
      op[trow + n * 16 + colid] = (bf16_t)(of[n][r] * inv);
  }
}

// ---------------------------------------------------------------------------
// host orchestration
// ---------------------------------------------------------------------------
extern "C" void kernel_launch(void* const* d_in, const int* in_sizes, int n_in,
                              void* d_out, int out_size, void* d_ws, size_t ws_size,
                              hipStream_t stream) {
  const float* x   = (const float*)d_in[0];
  const float* cs  = (const float*)d_in[1];
  const float* sn  = (const float*)d_in[2];
  const float* Wq  = (const float*)d_in[3];
  const float* Wk  = (const float*)d_in[4];
  const float* Wv  = (const float*)d_in[5];
  const float* Wo  = (const float*)d_in[6];
  const float* gq  = (const float*)d_in[7];
  const float* gk  = (const float*)d_in[8];
  float* out = (float*)d_out;

  char* p = (char*)d_ws;
  auto alloc = [&](size_t bytes) -> void* {
    void* r = (void*)p;
    p += (bytes + 255) & ~(size_t)255;
    return r;
  };

  const size_t nX  = (size_t)T_SEQ * D_MODEL;
  const size_t nQW = (size_t)D_MODEL * D_MODEL;
  const size_t nKW = (size_t)KV_D * D_MODEL;
  const size_t nKV = (size_t)T_SEQ * KV_D;

  bf16_t* xb  = (bf16_t*)alloc(nX  * 2);
  bf16_t* wqb = (bf16_t*)alloc(nQW * 2);
  bf16_t* wkb = (bf16_t*)alloc(nKW * 2);
  bf16_t* wvb = (bf16_t*)alloc(nKW * 2);
  bf16_t* wob = (bf16_t*)alloc(nQW * 2);
  float*  qf  = (float*) alloc(nX  * 4);
  float*  kf  = (float*) alloc(nKV * 4);
  bf16_t* qb  = (bf16_t*)alloc(nX  * 2);
  bf16_t* kb  = (bf16_t*)alloc(nKV * 2);
  bf16_t* vt  = (bf16_t*)alloc(nKV * 2);   // V^T: [1024, T]
  bf16_t* ab  = (bf16_t*)alloc(nX  * 2);

  const int B256 = 256;
  // ---- 1) casts
  cast_f32_bf16<<<(int)((nX  + 255) / 256), B256, 0, stream>>>(x,  xb,  (int)nX);
  cast_f32_bf16<<<(int)((nQW + 255) / 256), B256, 0, stream>>>(Wq, wqb, (int)nQW);
  cast_f32_bf16<<<(int)((nKW + 255) / 256), B256, 0, stream>>>(Wk, wkb, (int)nKW);
  cast_f32_bf16<<<(int)((nKW + 255) / 256), B256, 0, stream>>>(Wv, wvb, (int)nKW);
  cast_f32_bf16<<<(int)((nQW + 255) / 256), B256, 0, stream>>>(Wo, wob, (int)nQW);

  // ---- 2) projections (block tile 128x128)
  gemm_bf16_nt<float, false><<<dim3(T_SEQ / 128, D_MODEL / 128), B256, 0, stream>>>(
      xb, wqb, qf, T_SEQ, D_MODEL, D_MODEL);
  gemm_bf16_nt<float, false><<<dim3(T_SEQ / 128, KV_D / 128), B256, 0, stream>>>(
      xb, wkb, kf, T_SEQ, KV_D, D_MODEL);
  gemm_bf16_nt<bf16_t, true><<<dim3(T_SEQ / 128, KV_D / 128), B256, 0, stream>>>(
      xb, wvb, vt, T_SEQ, KV_D, D_MODEL);   // V straight to bf16, transposed

  // ---- 3) RMSNorm + RoPE
  rmsnorm_rope<<<(T_SEQ * NH_Q)  / 8, B256, 0, stream>>>(qf, gq, cs, sn, qb, NH_Q);
  rmsnorm_rope<<<(T_SEQ * NH_KV) / 8, B256, 0, stream>>>(kf, gk, cs, sn, kb, NH_KV);

  // ---- 4) causal flash attention
  flash_attn<<<(NH_Q * (T_SEQ / 16)) / 8, B256, 0, stream>>>(qb, kb, vt, ab);

  // ---- 5) output projection -> f32 d_out
  gemm_bf16_nt<float, false><<<dim3(T_SEQ / 128, D_MODEL / 128), B256, 0, stream>>>(
      ab, wob, out, T_SEQ, D_MODEL, D_MODEL);
}
